// CRFDecoder_76244259438905
// MI455X (gfx1250) — compile-verified, matrix operand order checked by reading, not run on hardware
//
#include <hip/hip_runtime.h>
#include <hip/hip_bf16.h>

// CRF NLL: B=128, T=2048, K=96
// loss = sum_b (logZ_b - score_b)

typedef __attribute__((ext_vector_type(16))) __bf16 v16bf;
typedef __attribute__((ext_vector_type(8)))  float  v8f;

#define Bb 128
#define Tt 2048
#define Kk 96
// Forward kernel: 8 blocks x 192 threads (6 waves). Each block owns 16 batch
// rows; wave w owns output tag-columns [w*16, w*16+16). K split into 3 k-tiles
// of 32 for v_wmma_f32_16x16x32_bf16.

__global__ __launch_bounds__(192) void crf_forward_kernel(
    const float* __restrict__ logits,          // [B][T][K]
    const unsigned char* __restrict__ mask,    // [B][T]
    const float* __restrict__ trans,           // [K][K]
    const float* __restrict__ start_t,         // [K]
    const float* __restrict__ end_t,           // [K]
    float* __restrict__ out)                   // [1] accumulates +logZ
{
    __shared__ float  s_alpha[16][Kk];                 // alpha, row-major
    __shared__ __align__(32) __bf16 s_A[3][32][16];    // A operand layout per k-tile
    __shared__ float  s_part[16][12];                  // partial row maxes
    __shared__ float  s_rmax[16];
    __shared__ float  s_maskf[16];

    const int tid  = threadIdx.x;
    const int lane = tid & 31;
    const int w    = tid >> 5;        // wave id == n-tile id (0..5)
    const int hl   = lane >> 4;       // half of wave
    const int l15  = lane & 15;
    const int b0   = blockIdx.x * 16; // first batch row of this block

    // ---- Resident B operand: E[k][n] = exp(trans[k][n]) for this wave's cols.
    // Element e of lane maps to k = (e>>3)*16 + hl*8 + (e&7) within k-tile.
    v16bf Bt[3];
    for (int kt = 0; kt < 3; ++kt) {
        for (int e = 0; e < 16; ++e) {
            int k = kt * 32 + ((e >> 3) << 4) + (hl << 3) + (e & 7);
            int n = w * 16 + l15;
            Bt[kt][e] = (__bf16)__expf(trans[k * Kk + n]);
        }
    }

    // ---- alpha0 = start + emissions(t=0); 1536 elems / 192 threads = 8 each
    for (int i = 0; i < 8; ++i) {
        int flat = tid * 8 + i;
        int r = flat / Kk, k = flat % Kk;
        s_alpha[r][k] = start_t[k] + logits[(size_t)(b0 + r) * Tt * Kk + k];
    }
    __syncthreads();

    const int colg = w * 16 + l15;   // this lane's global tag column

    for (int t = 1; t < Tt; ++t) {
        // Prefetch this step's emissions for our C-tile elements (hidden
        // behind phases A/B + barriers before use in phase D).
        float emit[8];
        for (int v = 0; v < 8; ++v) {
            int m = v + hl * 8;  // batch row within tile
            emit[v] = logits[(size_t)(b0 + m) * Tt * Kk + (size_t)t * Kk + colg];
        }

        // Phase A: partial row max (row = tid&15, segment of 8 cols = tid>>4)
        {
            int r = tid & 15, seg = tid >> 4;
            float mx = -1e30f;
            for (int i = 0; i < 8; ++i) mx = fmaxf(mx, s_alpha[r][seg * 8 + i]);
            s_part[r][seg] = mx;
            if (tid < 16) s_maskf[tid] = (float)mask[(size_t)(b0 + tid) * Tt + t];
        }
        __syncthreads();
        if (tid < 16) {
            float mx = -1e30f;
            for (int i = 0; i < 12; ++i) mx = fmaxf(mx, s_part[tid][i]);
            s_rmax[tid] = mx;
        }
        __syncthreads();

        // Phase B: p = exp(alpha - rowmax) stored straight in WMMA A layout
        for (int i = 0; i < 8; ++i) {
            int flat = tid * 8 + i;         // (kt, lane, e)
            int kt = flat >> 9;
            int l  = (flat >> 4) & 31;
            int e  = flat & 15;
            int r  = l & 15;
            int k  = kt * 32 + ((e >> 3) << 4) + ((l >> 4) << 3) + (e & 7);
            s_A[kt][l][e] = (__bf16)__expf(s_alpha[r][k] - s_rmax[r]);
        }
        __syncthreads();

        // Phase C: C = P x E  (3 chained bf16 WMMAs, f32 accumulate)
        v16bf A0 = *(const v16bf*)&s_A[0][lane][0];
        v16bf A1 = *(const v16bf*)&s_A[1][lane][0];
        v16bf A2 = *(const v16bf*)&s_A[2][lane][0];
        v8f c = {};
        c = __builtin_amdgcn_wmma_f32_16x16x32_bf16(false, A0, false, Bt[0],
                                                    (short)0, c, false, false);
        c = __builtin_amdgcn_wmma_f32_16x16x32_bf16(false, A1, false, Bt[1],
                                                    (short)0, c, false, false);
        c = __builtin_amdgcn_wmma_f32_16x16x32_bf16(false, A2, false, Bt[2],
                                                    (short)0, c, false, false);

        // Phase D: alpha' = log(C) + rowmax + emit, masked update
        for (int v = 0; v < 8; ++v) {
            int m = v + hl * 8;
            float newv = __logf(fmaxf(c[v], 1e-30f)) + s_rmax[m] + emit[v];
            float oldv = s_alpha[m][colg];
            s_alpha[m][colg] = (s_maskf[m] > 0.f) ? newv : oldv;
        }
        __syncthreads();
    }

    // Final: logZ_row = logsumexp(alpha + end_transitions)
    if (tid < 16) {
        float mx = -1e30f;
        for (int k = 0; k < Kk; ++k) mx = fmaxf(mx, s_alpha[tid][k] + end_t[k]);
        float s = 0.f;
        for (int k = 0; k < Kk; ++k) s += __expf(s_alpha[tid][k] + end_t[k] - mx);
        atomicAdd(out, mx + __logf(s));
    }
}

// Gold-path score: one wave per batch row; accumulates -score into out.
__global__ __launch_bounds__(32) void crf_score_kernel(
    const float* __restrict__ logits,
    const int* __restrict__ labels,
    const unsigned char* __restrict__ mask,
    const float* __restrict__ trans,
    const float* __restrict__ start_t,
    const float* __restrict__ end_t,
    float* __restrict__ out)
{
    int b = blockIdx.x;
    int lane = threadIdx.x;
    float acc = 0.f;
    int cnt = 0;
    for (int t = lane; t < Tt; t += 32) {
        int lab = labels[b * Tt + t];
        float mf = (float)mask[(size_t)b * Tt + t];
        cnt += (int)mask[(size_t)b * Tt + t];
        acc += mf * logits[((size_t)b * Tt + t) * Kk + lab];
        if (t > 0) {
            int labp = labels[b * Tt + t - 1];
            acc += mf * trans[labp * Kk + lab];
        }
    }
    for (int off = 16; off > 0; off >>= 1) {
        acc += __shfl_down(acc, off, 32);
        cnt += __shfl_down(cnt, off, 32);
    }
    if (lane == 0) {
        int lab0 = labels[b * Tt];
        int last = labels[b * Tt + (cnt > 0 ? cnt - 1 : 0)];
        float sc = start_t[lab0] + acc + end_t[last];
        atomicAdd(out, -sc);
    }
}

__global__ void crf_init_out(float* out) {
    if (threadIdx.x == 0 && blockIdx.x == 0) out[0] = 0.f;
}

extern "C" void kernel_launch(void* const* d_in, const int* in_sizes, int n_in,
                              void* d_out, int out_size, void* d_ws, size_t ws_size,
                              hipStream_t stream) {
    (void)in_sizes; (void)n_in; (void)out_size; (void)d_ws; (void)ws_size;
    const float*         logits  = (const float*)d_in[0];
    const int*           labels  = (const int*)d_in[1];
    const unsigned char* mask    = (const unsigned char*)d_in[2];
    const float*         trans   = (const float*)d_in[3];
    const float*         start_t = (const float*)d_in[4];
    const float*         end_t   = (const float*)d_in[5];
    float* out = (float*)d_out;

    crf_init_out<<<1, 32, 0, stream>>>(out);
    crf_forward_kernel<<<Bb / 16, 192, 0, stream>>>(logits, mask, trans,
                                                    start_t, end_t, out);
    crf_score_kernel<<<Bb, 32, 0, stream>>>(logits, labels, mask, trans,
                                            start_t, end_t, out);
}
